// ForceTokenizer_62311385530809
// MI455X (gfx1250) — matrix operand order repaired and checked by reading, not compiled, not run on hardware
//
#include <hip/hip_runtime.h>
#include <hip/hip_bf16.h>

// ---------------------------------------------------------------------------
// ForceTokenizer for MI455X (gfx1250, wave32, WMMA)
//
// Activations: NHWC with a 1-pixel ZERO HALO [B][H+2][W+2][C] f16 -> 'same'
// padding via memory zeros; no border logic in the hot loop.
// K ordering k = (r*3+s)*C + c; rs-outer (fully unrolled -> tap offsets are
// compile-time immediates folded into the load ioffset) / channel-inner.
// One base address register pair per pixel; zero address VALU in the K loop.
//
// conv2 uses MBLK=4: each wave owns 64M x 32N (8 accumulators); the 4 B-chunk
// loads per K-step feed 8 WMMAs (1.5 loads/wmma) and each h1 byte is pulled
// by a single block -> B HBM traffic ~0.3 GB, matrix-pipe bound.
//
//   conv1: 6->256  s1 + SiLU   C padded to 8 (3 K-steps, const-select taps)
//   conv2: 256->256 s2 + SiLU  MBLK=4
//   conv3: 256->64  s2         f32 out [pixel][64] (VQ layout)
//   VQ:    argmin over 512 codes, dim 64 + commitment loss
// ---------------------------------------------------------------------------

typedef __attribute__((ext_vector_type(16))) _Float16 v16h;
typedef __attribute__((ext_vector_type(8)))  _Float16 v8h;
typedef __attribute__((ext_vector_type(8)))  float    v8f;
typedef __attribute__((ext_vector_type(4)))  float    v4f;

union HF { v16h v; v8h h[2]; };

// -------------------------------- weight packing ---------------------------
__global__ __launch_bounds__(256)
void pack_weights_kernel(const float* __restrict__ w, _Float16* __restrict__ wp,
                         int OC, int IC, int ICP, int KPAD) {
    int gid = blockIdx.x * 256 + threadIdx.x;
    if (gid >= OC * KPAD) return;
    int oc = gid / KPAD;
    int k  = gid - oc * KPAD;
    int rs = k / ICP;
    int ic = k - rs * ICP;
    float v = (rs < 9 && ic < IC) ? w[((size_t)oc * IC + ic) * 9 + rs] : 0.0f;
    wp[gid] = (_Float16)v;
}

// --------------- input: NCHW f32 -> halo'd NHWC(8) f16 ---------------------
__global__ __launch_bounds__(256)
void convert_input_kernel(const float* __restrict__ x, _Float16* __restrict__ y) {
    int gid = blockIdx.x * 256 + threadIdx.x;   // over 8*258*258*8
    if (gid >= 8 * 258 * 258 * 8) return;
    int c   = gid & 7;
    int pix = gid >> 3;
    int b   = pix / (258 * 258);
    int rem = pix - b * (258 * 258);
    int yy  = rem / 258;
    int xx  = rem - yy * 258;
    float v = 0.0f;
    if (c < 6 && yy >= 1 && yy <= 256 && xx >= 1 && xx <= 256)
        v = x[(((size_t)b * 6 + c) << 16) + ((yy - 1) << 8) + (xx - 1)];
    y[gid] = (_Float16)v;
}

// --------------- zero the 1-pixel halo of an NHWC buffer -------------------
__global__ __launch_bounds__(256)
void zero_halo_kernel(_Float16* __restrict__ buf, int B, int H, int W, int C) {
    int Wh = W + 2, Hh = H + 2;
    int P  = 2 * Wh + 2 * H;
    int vecs = C >> 3;
    int total = B * P * vecs;
    int gid = blockIdx.x * 256 + threadIdx.x;
    if (gid >= total) return;
    int v     = gid % vecs;
    int pixid = gid / vecs;
    int b = pixid / P;
    int p = pixid - b * P;
    int yy, xx;
    if (p < Wh)           { yy = 0;      xx = p; }
    else if (p < 2 * Wh)  { yy = Hh - 1; xx = p - Wh; }
    else { int q = p - 2 * Wh; yy = 1 + (q >> 1); xx = (q & 1) ? (Wh - 1) : 0; }
    v8h z = {};
    *(v8h*)(buf + ((size_t)(b * Hh + yy) * Wh + xx) * C + v * 8) = z;
}

// -------------------------------- conv as implicit GEMM --------------------
// 256 threads = 8 waves; wave w: M-group = (w>>1)*16*MBLK, N-subtile = w&1.
// Wave tile: (16*MBLK) M x 32 N -> MBLK x 2 accumulators.
// Block tile: (64*MBLK) M x 64 N.  grid.x = N/64, grid.y = OC/(64*MBLK).
template<int ICP, int LOG_ICP, int OC, int H, int W, int STRIDE,
         bool SILU, bool ZOUT, int MBLK>
__global__ __launch_bounds__(256)
void conv3x3_wmma_kernel(const _Float16* __restrict__ X,   // halo NHWC
                         const _Float16* __restrict__ Wp,  // [OC][KPAD]
                         const float*    __restrict__ bias,
                         _Float16*       __restrict__ Y,   // halo NHWC f16 out
                         float*          __restrict__ Z) { // f32 [n][OC]
    constexpr int Hh = H + 2, Wh = W + 2;
    constexpr int OH = H / STRIDE, OW = W / STRIDE;
    constexpr int OWh = OW + 2;
    constexpr int KPAD = (ICP >= 32) ? 9 * ICP : 96;

    const int tid    = threadIdx.x;
    const int lane   = tid & 31;
    const int wave   = tid >> 5;
    const int nt     = wave & 1;
    const int lanehi = (lane >> 4) & 1;
    const int l15    = lane & 15;

    const int mbase = blockIdx.y * (64 * MBLK) + (wave >> 1) * (16 * MBLK);
    const int nbase = blockIdx.x * 64 + nt * 32;

    const int koffA = lanehi * 16;
    const int koffB = lanehi * 16;
    // single A base; M-subtile j at immediate offset j*16*KPAD
    const _Float16* wk = Wp + (size_t)(mbase + l15) * KPAD + koffA;

    // ---- per-lane output pixels ----
    const int n0 = nbase + l15;
    const int n1 = n0 + 16;
    int b0, oy0, ox0, b1, oy1, ox1;
    {
        b0 = n0 / (OH * OW);                    // power-of-two -> shifts
        int rem = n0 - b0 * (OH * OW);
        oy0 = rem / OW; ox0 = rem - oy0 * OW;
        b1 = n1 / (OH * OW);
        rem = n1 - b1 * (OH * OW);
        oy1 = rem / OW; ox1 = rem - oy1 * OW;
    }
    // single B base per pixel (center tap); tap/channel offsets are immediates
    const _Float16* xb0 = X +
        (((size_t)((b0 * Hh + oy0 * STRIDE + 1) * Wh + ox0 * STRIDE + 1)) << LOG_ICP) + koffB;
    const _Float16* xb1 = X +
        (((size_t)((b1 * Hh + oy1 * STRIDE + 1) * Wh + ox1 * STRIDE + 1)) << LOG_ICP) + koffB;

    v8f acc[MBLK][2];
    #pragma unroll
    for (int j = 0; j < MBLK; ++j) { acc[j][0] = {}; acc[j][1] = {}; }

    if constexpr (ICP >= 32) {
        #pragma unroll
        for (int rs = 0; rs < 9; ++rs) {
            const int doff = ((rs / 3 - 1) * Wh + (rs - (rs / 3) * 3 - 1)) << LOG_ICP;
            #pragma unroll
            for (int icb = 0; icb < ICP; icb += 32) {
                HF f0, f1;
                f0.h[0] = *(const v8h*)(xb0 + doff + icb);
                f0.h[1] = *(const v8h*)(xb0 + doff + icb + 8);
                f1.h[0] = *(const v8h*)(xb1 + doff + icb);
                f1.h[1] = *(const v8h*)(xb1 + doff + icb + 8);
                #pragma unroll
                for (int j = 0; j < MBLK; ++j) {
                    HF a;
                    a.h[0] = *(const v8h*)(wk + j * 16 * KPAD + rs * ICP + icb);
                    a.h[1] = *(const v8h*)(wk + j * 16 * KPAD + rs * ICP + icb + 8);
                    acc[j][0] = __builtin_amdgcn_wmma_f32_16x16x32_f16(
                        false, a.v, false, f0.v, (short)0, acc[j][0], false, false);
                    acc[j][1] = __builtin_amdgcn_wmma_f32_16x16x32_f16(
                        false, a.v, false, f1.v, (short)0, acc[j][1], false, false);
                }
            }
        }
    } else {
        // conv1 (ICP==8): chunk rs = kb/8 + j + 2*lanehi.  Both candidates are
        // compile-time -> one cndmask between two constant offsets; rs>=9
        // guard folds (only the tail step has an invalid variant).
        auto chunkB = [&](int rs0, const _Float16* xb) -> v8h {
            const int  rsA = rs0, rsB = rs0 + 2;          // consts after unroll
            const bool okA = rsA < 9, okB = rsB < 9;
            const int  dA  = okA ? (((rsA / 3 - 1) * Wh + (rsA % 3 - 1)) << 3) : 0;
            const int  dB  = okB ? (((rsB / 3 - 1) * Wh + (rsB % 3 - 1)) << 3) : 0;
            const int  d   = lanehi ? dB : dA;
            const bool ok  = lanehi ? okB : okA;
            v8h v = *(const v8h*)(xb - koffB + d);        // koffB==2*koffA? no: undo B off
            v8h z = {};
            return ok ? v : z;
        };
        #pragma unroll
        for (int kb = 0; kb < KPAD; kb += 32) {
            HF a, f0, f1;
            a.h[0]  = *(const v8h*)(wk + kb);
            a.h[1]  = *(const v8h*)(wk + kb + 8);
            f0.h[0] = chunkB(kb / 8,     xb0);
            f0.h[1] = chunkB(kb / 8 + 1, xb0);
            f1.h[0] = chunkB(kb / 8,     xb1);
            f1.h[1] = chunkB(kb / 8 + 1, xb1);
            acc[0][0] = __builtin_amdgcn_wmma_f32_16x16x32_f16(
                false, a.v, false, f0.v, (short)0, acc[0][0], false, false);
            acc[0][1] = __builtin_amdgcn_wmma_f32_16x16x32_f16(
                false, a.v, false, f1.v, (short)0, acc[0][1], false, false);
        }
    }

    // ---- epilogue: bias (+SiLU); vectorized stores ----
    #pragma unroll
    for (int j = 0; j < MBLK; ++j) {
        const int m0 = mbase + j * 16 + lanehi * 8;
        float bz[8];
        #pragma unroll
        for (int r = 0; r < 8; ++r) bz[r] = bias[m0 + r];

        if constexpr (ZOUT) {
            v4f a0 = {acc[j][0][0] + bz[0], acc[j][0][1] + bz[1],
                      acc[j][0][2] + bz[2], acc[j][0][3] + bz[3]};
            v4f a1 = {acc[j][0][4] + bz[4], acc[j][0][5] + bz[5],
                      acc[j][0][6] + bz[6], acc[j][0][7] + bz[7]};
            v4f c0 = {acc[j][1][0] + bz[0], acc[j][1][1] + bz[1],
                      acc[j][1][2] + bz[2], acc[j][1][3] + bz[3]};
            v4f c1 = {acc[j][1][4] + bz[4], acc[j][1][5] + bz[5],
                      acc[j][1][6] + bz[6], acc[j][1][7] + bz[7]};
            *(v4f*)(Z + (size_t)n0 * OC + m0)     = a0;
            *(v4f*)(Z + (size_t)n0 * OC + m0 + 4) = a1;
            *(v4f*)(Z + (size_t)n1 * OC + m0)     = c0;
            *(v4f*)(Z + (size_t)n1 * OC + m0 + 4) = c1;
        } else {
            v8h o0, o1;
            #pragma unroll
            for (int r = 0; r < 8; ++r) {
                float y0 = acc[j][0][r] + bz[r];
                float y1 = acc[j][1][r] + bz[r];
                if (SILU) {
                    y0 = y0 / (1.0f + __expf(-y0));
                    y1 = y1 / (1.0f + __expf(-y1));
                }
                o0[r] = (_Float16)y0;
                o1[r] = (_Float16)y1;
            }
            size_t off0 = ((size_t)(b0 * (OH + 2) + oy0 + 1) * OWh + ox0 + 1) * OC + m0;
            size_t off1 = ((size_t)(b1 * (OH + 2) + oy1 + 1) * OWh + ox1 + 1) * OC + m0;
            *(v8h*)(Y + off0) = o0;              // global_store_b128
            *(v8h*)(Y + off1) = o1;
        }
    }
}

// -------------------------------- VQ: nearest code + loss ------------------
__global__ __launch_bounds__(256)
void vq_kernel(const float* __restrict__ Z, const float* __restrict__ CB,
               float* __restrict__ tokens, float* __restrict__ loss_acc) {
    __shared__ float scb[64 * 64];
    __shared__ float sred[256];

    const int gid = blockIdx.x * 256 + threadIdx.x;
    float zr[64];
    #pragma unroll
    for (int d = 0; d < 64; ++d) zr[d] = Z[(size_t)gid * 64 + d];

    float best = 3.4e38f;
    int   bi   = 0;
    for (int c0 = 0; c0 < 512; c0 += 64) {
        __syncthreads();
        #pragma unroll
        for (int i = 0; i < 16; ++i) {
            int idx = threadIdx.x + i * 256;
            scb[idx] = CB[(size_t)c0 * 64 + idx];
        }
        __syncthreads();
        for (int j = 0; j < 64; ++j) {
            float d2 = 0.0f;
            #pragma unroll
            for (int d = 0; d < 64; ++d) {
                float diff = zr[d] - scb[j * 64 + d];
                d2 = fmaf(diff, diff, d2);
            }
            if (d2 < best) { best = d2; bi = c0 + j; }
        }
    }
    tokens[gid] = (float)bi;

    __syncthreads();
    sred[threadIdx.x] = best;
    __syncthreads();
    for (int off = 128; off > 0; off >>= 1) {
        if (threadIdx.x < (unsigned)off) sred[threadIdx.x] += sred[threadIdx.x + off];
        __syncthreads();
    }
    if (threadIdx.x == 0) atomicAdd(loss_acc, sred[0]);
}

__global__ void init_loss_kernel(float* loss_acc) {
    if (threadIdx.x == 0) loss_acc[0] = 0.0f;
}

__global__ void finalize_loss_kernel(const float* loss_acc, float* out) {
    if (threadIdx.x == 0)
        out[0] = loss_acc[0] * (1.25f / 2097152.0f);  // 1.25 * sum / (N*64)
}

// ---------------------------------------------------------------------------
extern "C" void kernel_launch(void* const* d_in, const int* in_sizes, int n_in,
                              void* d_out, int out_size, void* d_ws, size_t ws_size,
                              hipStream_t stream) {
    const float* force = (const float*)d_in[0];
    const float* w1    = (const float*)d_in[1];
    const float* b1    = (const float*)d_in[2];
    const float* w2    = (const float*)d_in[3];
    const float* b2    = (const float*)d_in[4];
    const float* w3    = (const float*)d_in[5];
    const float* b3    = (const float*)d_in[6];
    const float* cb    = (const float*)d_in[7];

    char* ws = (char*)d_ws;
    size_t off = 0;
    auto carve = [&](size_t bytes) -> char* {
        char* p = ws + off;
        off += (bytes + 255) & ~(size_t)255;
        return p;
    };
    _Float16* xh   = (_Float16*)carve((size_t)8 * 258 * 258 * 8 * 2);
    _Float16* w1p  = (_Float16*)carve((size_t)256 * 96 * 2);
    _Float16* w2p  = (_Float16*)carve((size_t)256 * 2304 * 2);
    _Float16* w3p  = (_Float16*)carve((size_t)64 * 2304 * 2);
    _Float16* h1   = (_Float16*)carve((size_t)8 * 258 * 258 * 256 * 2);
    _Float16* h2   = (_Float16*)carve((size_t)8 * 130 * 130 * 256 * 2);
    float*    zf   = (float*)   carve((size_t)32768 * 64 * 4);
    float*    loss = (float*)   carve(256);

    float* tokens   = (float*)d_out;
    float* loss_out = (float*)d_out + 32768;

    // ---- prep ----
    pack_weights_kernel<<<(256 * 96   + 255) / 256, 256, 0, stream>>>(w1, w1p, 256, 6,   8,   96);
    pack_weights_kernel<<<(256 * 2304 + 255) / 256, 256, 0, stream>>>(w2, w2p, 256, 256, 256, 2304);
    pack_weights_kernel<<<(64  * 2304 + 255) / 256, 256, 0, stream>>>(w3, w3p, 64,  256, 256, 2304);
    convert_input_kernel<<<(8 * 258 * 258 * 8 + 255) / 256, 256, 0, stream>>>(force, xh);
    {
        int t1 = 8 * (2 * 258 + 2 * 256) * (256 / 8);
        zero_halo_kernel<<<(t1 + 255) / 256, 256, 0, stream>>>(h1, 8, 256, 256, 256);
        int t2 = 8 * (2 * 130 + 2 * 128) * (256 / 8);
        zero_halo_kernel<<<(t2 + 255) / 256, 256, 0, stream>>>(h2, 8, 128, 128, 256);
    }
    init_loss_kernel<<<1, 32, 0, stream>>>(loss);

    // ---- encoder ----
    // conv1: N = 524288, MBLK=1 -> grid (8192, 4)
    conv3x3_wmma_kernel<8, 3, 256, 256, 256, 1, true, false, 1>
        <<<dim3(8192, 4), 256, 0, stream>>>(xh, w1p, b1, h1, nullptr);
    // conv2: N = 131072, MBLK=4 (block covers all 256 OC) -> grid (2048, 1)
    conv3x3_wmma_kernel<256, 8, 256, 256, 256, 2, true, false, 4>
        <<<dim3(2048, 1), 256, 0, stream>>>(h1, w2p, b2, h2, nullptr);
    // conv3: N = 32768, MBLK=1 -> grid (512, 1)
    conv3x3_wmma_kernel<256, 8, 64, 128, 128, 2, false, true, 1>
        <<<dim3(512, 1), 256, 0, stream>>>(h2, w3p, b3, nullptr, zf);

    // ---- VQ + loss ----
    vq_kernel<<<128, 256, 0, stream>>>(zf, cb, tokens, loss);
    finalize_loss_kernel<<<1, 32, 0, stream>>>(loss, loss_out);

    (void)in_sizes; (void)n_in; (void)out_size; (void)ws_size;
}
